// AdaptiveComputationTime_43508018709042
// MI455X (gfx1250) — matrix operand order repaired and checked by reading, not compiled, not run on hardware
//
#include <hip/hip_runtime.h>
#include <hip/hip_bf16.h>
#include <math.h>

// ---------------------------------------------------------------------------
// Types for CDNA5 WMMA (gfx1250, wave32)
// ---------------------------------------------------------------------------
typedef __attribute__((ext_vector_type(16))) __bf16         v16bf;
typedef __attribute__((ext_vector_type(8)))  float          v8f;
typedef __attribute__((ext_vector_type(8)))  unsigned short ush8;
typedef __attribute__((ext_vector_type(16))) unsigned short ush16;
typedef int b128_t __attribute__((vector_size(16)));   // pointee type of async-LDS builtins

union Frag16 { ush8 h[2]; ush16 w; };

#define B_SZ 1024
#define I_SZ 2048
#define H_SZ 2048
#define G4   (4 * H_SZ)      // 8192
#define NSTEPS 10
#define THRESH 0.99f

// Async global->LDS (CDNA5 GLOBAL_LOAD_ASYNC_TO_LDS_*, tracked by ASYNCcnt)
#if defined(__has_builtin)
#  if __has_builtin(__builtin_amdgcn_global_load_async_to_lds_b128) && \
      __has_builtin(__builtin_amdgcn_s_wait_asynccnt)
#    define USE_ASYNC_LDS 1
#  endif
#endif
#ifndef USE_ASYNC_LDS
#  define USE_ASYNC_LDS 0
#endif

__device__ __forceinline__ void cp16_async(const unsigned short* g, unsigned short* l) {
#if USE_ASYNC_LDS
  __builtin_amdgcn_global_load_async_to_lds_b128(
      (__attribute__((address_space(1))) b128_t*)(g),
      (__attribute__((address_space(3))) b128_t*)(l), 0, 0);
#else
  (void)g; (void)l;
#endif
}

__device__ __forceinline__ unsigned short f2bf(float f) {
  unsigned int u = __builtin_bit_cast(unsigned int, f);
  unsigned int r = u + 0x7FFFu + ((u >> 16) & 1u);   // round-to-nearest-even
  return (unsigned short)(r >> 16);
}
__device__ __forceinline__ float sigm(float x) { return 1.0f / (1.0f + __expf(-x)); }

// one A-frag x 4 B-frags -> 4 WMMA accumulators (i,f,g,o)
__device__ __forceinline__ void mma4(const unsigned short* Arow,
                                     const unsigned short* B0, const unsigned short* B1,
                                     const unsigned short* B2, const unsigned short* B3,
                                     int koa, int kob,
                                     v8f& ai, v8f& af, v8f& ag, v8f& ao) {
  Frag16 a;  a.h[0]  = *(const ush8*)(Arow + koa); a.h[1]  = *(const ush8*)(Arow + 8 + koa);
  Frag16 b0; b0.h[0] = *(const ush8*)(B0 + kob);   b0.h[1] = *(const ush8*)(B0 + kob + 8);
  Frag16 b1; b1.h[0] = *(const ush8*)(B1 + kob);   b1.h[1] = *(const ush8*)(B1 + kob + 8);
  Frag16 b2; b2.h[0] = *(const ush8*)(B2 + kob);   b2.h[1] = *(const ush8*)(B2 + kob + 8);
  Frag16 b3; b3.h[0] = *(const ush8*)(B3 + kob);   b3.h[1] = *(const ush8*)(B3 + kob + 8);
  v16bf av = __builtin_bit_cast(v16bf, a.w);
  ai = __builtin_amdgcn_wmma_f32_16x16x32_bf16(false, av, false, __builtin_bit_cast(v16bf, b0.w), (short)0, ai, false, false);
  af = __builtin_amdgcn_wmma_f32_16x16x32_bf16(false, av, false, __builtin_bit_cast(v16bf, b1.w), (short)0, af, false, false);
  ag = __builtin_amdgcn_wmma_f32_16x16x32_bf16(false, av, false, __builtin_bit_cast(v16bf, b2.w), (short)0, ag, false, false);
  ao = __builtin_amdgcn_wmma_f32_16x16x32_bf16(false, av, false, __builtin_bit_cast(v16bf, b3.w), (short)0, ao, false, false);
}

// ---------------------------------------------------------------------------
// fp32 -> bf16 conversion (4 elems/thread, b128 loads)
// ---------------------------------------------------------------------------
struct U4 { unsigned short a, b, c, d; };
__global__ void cvt_bf16(const float* __restrict__ src, unsigned short* __restrict__ dst, int n4) {
  int i = blockIdx.x * blockDim.x + threadIdx.x;
  if (i < n4) {
    float4 f = ((const float4*)src)[i];
    U4 o; o.a = f2bf(f.x); o.b = f2bf(f.y); o.c = f2bf(f.z); o.d = f2bf(f.w);
    ((U4*)dst)[i] = o;
  }
}

// ---------------------------------------------------------------------------
// state init
// ---------------------------------------------------------------------------
__global__ void init_state(float* h, float* c, unsigned short* hbf0, float* hsum,
                           float* hp, float* rem, float* nup, int* flags) {
  int i = blockIdx.x * blockDim.x + threadIdx.x;
  if (i < B_SZ * H_SZ) { h[i] = 0.f; c[i] = 0.f; hbf0[i] = 0; hsum[i] = 0.f; }
  if (i < B_SZ) { hp[i] = 0.f; rem[i] = 0.f; nup[i] = 0.f; }
  if (i == 0) { flags[0] = 1; flags[1] = 0; }
}

// ---------------------------------------------------------------------------
// xg = x @ W_ih^T + (b_ih + b_hh)    [1024 x 8192], K = 2048, bf16 WMMA
// block: 256 thr (8 waves), tile M=64 x N=64; wave (wm 0..3, wn 0..1) -> 16x32
// Double-buffered LDS; async global->LDS when available.
// ---------------------------------------------------------------------------
__global__ __launch_bounds__(256)
void gemm_xg(const unsigned short* __restrict__ A,   // x_bf   [1024 x 2048]
             const unsigned short* __restrict__ W,   // Wih_bf [8192 x 2048]
             const float* __restrict__ b_ih, const float* __restrict__ b_hh,
             float* __restrict__ xg) {
#if USE_ASYNC_LDS
  __shared__ __align__(16) unsigned short As[2][64][32];
  __shared__ __align__(16) unsigned short Bs[2][64][32];
#else
  __shared__ __align__(16) unsigned short As[1][64][32];
  __shared__ __align__(16) unsigned short Bs[1][64][32];
#endif
  const int m0 = blockIdx.y * 64;
  const int n0 = blockIdx.x * 64;
  const int tid  = threadIdx.x;
  const int lane = tid & 31;
  const int wave = tid >> 5;
  const int wm = wave >> 1;
  const int wn = wave & 1;
  const int arow = tid >> 2, akc = tid & 3;

  v8f acc0 = {}; v8f acc1 = {};
  const int koa = (lane & 16) ? 16 : 0;
  const int kob = (lane & 16) ? 16 : 0;
  const int ar  = wm * 16 + (lane & 15);
  const int bc0 = wn * 32 + (lane & 15);
  const int bc1 = bc0 + 16;
  const unsigned short* gA = &A[(size_t)(m0 + arow) * I_SZ + akc * 8];
  const unsigned short* gB = &W[(size_t)(n0 + arow) * I_SZ + akc * 8];

#if USE_ASYNC_LDS
  auto issue = [&](int k0, int buf) {
    cp16_async(gA + k0, &As[buf][arow][akc * 8]);
    cp16_async(gB + k0, &Bs[buf][arow][akc * 8]);
  };
  issue(0, 0);
  for (int t = 0; t < I_SZ / 32; ++t) {
    const int cur = t & 1;
    if (t + 1 < I_SZ / 32) {
      issue((t + 1) * 32, cur ^ 1);
      __builtin_amdgcn_s_wait_asynccnt(2);   // tile t landed; tile t+1 in flight
    } else {
      __builtin_amdgcn_s_wait_asynccnt(0);
    }
    __syncthreads();
    {
      Frag16 a;  a.h[0]  = *(const ush8*)&As[cur][ar][koa];   a.h[1]  = *(const ush8*)&As[cur][ar][8 + koa];
      Frag16 b0; b0.h[0] = *(const ush8*)&Bs[cur][bc0][kob];  b0.h[1] = *(const ush8*)&Bs[cur][bc0][kob + 8];
      Frag16 b1; b1.h[0] = *(const ush8*)&Bs[cur][bc1][kob];  b1.h[1] = *(const ush8*)&Bs[cur][bc1][kob + 8];
      v16bf av = __builtin_bit_cast(v16bf, a.w);
      acc0 = __builtin_amdgcn_wmma_f32_16x16x32_bf16(false, av, false, __builtin_bit_cast(v16bf, b0.w), (short)0, acc0, false, false);
      acc1 = __builtin_amdgcn_wmma_f32_16x16x32_bf16(false, av, false, __builtin_bit_cast(v16bf, b1.w), (short)0, acc1, false, false);
    }
    __syncthreads();   // readers done before next issue overwrites this buffer
  }
#else
  for (int k0 = 0; k0 < I_SZ; k0 += 32) {
    *(ush8*)&As[0][arow][akc * 8] = *(const ush8*)(gA + k0);
    *(ush8*)&Bs[0][arow][akc * 8] = *(const ush8*)(gB + k0);
    __syncthreads();
    Frag16 a;  a.h[0]  = *(const ush8*)&As[0][ar][koa];  a.h[1]  = *(const ush8*)&As[0][ar][8 + koa];
    Frag16 b0; b0.h[0] = *(const ush8*)&Bs[0][bc0][kob]; b0.h[1] = *(const ush8*)&Bs[0][bc0][kob + 8];
    Frag16 b1; b1.h[0] = *(const ush8*)&Bs[0][bc1][kob]; b1.h[1] = *(const ush8*)&Bs[0][bc1][kob + 8];
    v16bf av = __builtin_bit_cast(v16bf, a.w);
    acc0 = __builtin_amdgcn_wmma_f32_16x16x32_bf16(false, av, false, __builtin_bit_cast(v16bf, b0.w), (short)0, acc0, false, false);
    acc1 = __builtin_amdgcn_wmma_f32_16x16x32_bf16(false, av, false, __builtin_bit_cast(v16bf, b1.w), (short)0, acc1, false, false);
    __syncthreads();
  }
#endif

  const int roff = (lane & 16) ? 8 : 0;
  const int col0 = n0 + wn * 32 + (lane & 15);
  const int col1 = col0 + 16;
  const float bias0 = b_ih[col0] + b_hh[col0];
  const float bias1 = b_ih[col1] + b_hh[col1];
  for (int v = 0; v < 8; ++v) {
    int row = m0 + wm * 16 + roff + v;
    xg[(size_t)row * G4 + col0] = acc0[v] + bias0;
    xg[(size_t)row * G4 + col1] = acc1[v] + bias1;
  }
}

// ---------------------------------------------------------------------------
// One ACT step, fully fused: gates = xg + h @ W_hh^T, then LSTM pointwise.
// Block owns 64(row) x 32(hidden); all four gate tiles meet in registers.
// Double-buffered LDS with async global->LDS copies.
// ---------------------------------------------------------------------------
__global__ __launch_bounds__(256)
void lstm_step(const unsigned short* __restrict__ hbf_in,  // [1024 x 2048]
               const unsigned short* __restrict__ Whh,     // [8192 x 2048] bf16
               const float* __restrict__ xg,               // [1024 x 8192]
               float* __restrict__ c, float* __restrict__ h,
               unsigned short* __restrict__ hbf_out,
               float* __restrict__ hsum,
               const int* __restrict__ flags) {
  if (flags[0] == 0) return;   // global scalar early-exit (uniform)

#if USE_ASYNC_LDS
  __shared__ __align__(16) unsigned short As[2][64][32];
  __shared__ __align__(16) unsigned short Bs[2][4][32][32];
#else
  __shared__ __align__(16) unsigned short As[1][64][32];
  __shared__ __align__(16) unsigned short Bs[1][4][32][32];
#endif
  const int m0 = blockIdx.y * 64;   // batch rows
  const int n0 = blockIdx.x * 32;   // hidden cols
  const int tid  = threadIdx.x;
  const int lane = tid & 31;
  const int wave = tid >> 5;
  const int wm = wave >> 1;
  const int wn = wave & 1;

  v8f acc_i = {}, acc_f = {}, acc_g = {}, acc_o = {};
  const int koa = (lane & 16) ? 16 : 0;
  const int kob = (lane & 16) ? 16 : 0;
  const int ar  = wm * 16 + (lane & 15);
  const int bc  = wn * 16 + (lane & 15);
  const int arow = tid >> 2, akc = tid & 3;
  // B staging: 512 chunks (4 gates x 32 cols x 4 kchunks), 2 per thread
  const int g1 = tid >> 7,         n1 = (tid >> 2) & 31,         kc1 = tid & 3;
  const int g2 = (tid + 256) >> 7, n2 = ((tid + 256) >> 2) & 31, kc2 = tid & 3;
  const unsigned short* gA  = &hbf_in[(size_t)(m0 + arow) * H_SZ + akc * 8];
  const unsigned short* gB1 = &Whh[(size_t)(g1 * H_SZ + n0 + n1) * H_SZ + kc1 * 8];
  const unsigned short* gB2 = &Whh[(size_t)(g2 * H_SZ + n0 + n2) * H_SZ + kc2 * 8];

#if USE_ASYNC_LDS
  auto issue = [&](int k0, int buf) {
    cp16_async(gA  + k0, &As[buf][arow][akc * 8]);
    cp16_async(gB1 + k0, &Bs[buf][g1][n1][kc1 * 8]);
    cp16_async(gB2 + k0, &Bs[buf][g2][n2][kc2 * 8]);
  };
  issue(0, 0);
  for (int t = 0; t < H_SZ / 32; ++t) {
    const int cur = t & 1;
    if (t + 1 < H_SZ / 32) {
      issue((t + 1) * 32, cur ^ 1);
      __builtin_amdgcn_s_wait_asynccnt(3);   // current tile landed, next in flight
    } else {
      __builtin_amdgcn_s_wait_asynccnt(0);
    }
    __syncthreads();
    mma4(&As[cur][ar][0], &Bs[cur][0][bc][0], &Bs[cur][1][bc][0],
         &Bs[cur][2][bc][0], &Bs[cur][3][bc][0], koa, kob, acc_i, acc_f, acc_g, acc_o);
    __syncthreads();
  }
#else
  for (int k0 = 0; k0 < H_SZ; k0 += 32) {
    *(ush8*)&As[0][arow][akc * 8]   = *(const ush8*)(gA + k0);
    *(ush8*)&Bs[0][g1][n1][kc1 * 8] = *(const ush8*)(gB1 + k0);
    *(ush8*)&Bs[0][g2][n2][kc2 * 8] = *(const ush8*)(gB2 + k0);
    if (k0 + 32 < H_SZ) __builtin_prefetch(gA + k0 + 32, 0, 1);
    __syncthreads();
    mma4(&As[0][ar][0], &Bs[0][0][bc][0], &Bs[0][1][bc][0],
         &Bs[0][2][bc][0], &Bs[0][3][bc][0], koa, kob, acc_i, acc_f, acc_g, acc_o);
    __syncthreads();
  }
#endif

  // fused LSTM pointwise on the 16x16 patch this wave owns
  const int roff = (lane & 16) ? 8 : 0;
  const int col  = n0 + wn * 16 + (lane & 15);
  for (int v = 0; v < 8; ++v) {
    int row = m0 + wm * 16 + roff + v;
    size_t xb = (size_t)row * G4 + col;
    float ig = acc_i[v] + xg[xb];
    float fg = acc_f[v] + xg[xb + H_SZ];
    float gg = acc_g[v] + xg[xb + 2 * H_SZ];
    float og = acc_o[v] + xg[xb + 3 * H_SZ];
    size_t hb = (size_t)row * H_SZ + col;
    float cn = sigm(fg) * c[hb] + sigm(ig) * tanhf(gg);
    float hn = sigm(og) * tanhf(cn);
    c[hb] = cn;
    h[hb] = hn;
    hbf_out[hb] = f2bf(hn);
    hsum[hb] += hn;
  }
}

// ---------------------------------------------------------------------------
// Halting head: y = sigmoid(h_new . W_halt + b_halt); update hp/rem/nup;
// flag any row with new hp <= THRESH. One 256-thread block per batch row.
// ---------------------------------------------------------------------------
__global__ __launch_bounds__(256)
void halt_update(const float* __restrict__ h, const float* __restrict__ W_halt,
                 const float* __restrict__ b_halt,
                 float* hp, float* rem, float* nup, int* flags) {
  if (flags[0] == 0) return;
  __shared__ float red[8];
  const int b = blockIdx.x;
  const int lane = threadIdx.x & 31, wave = threadIdx.x >> 5;
  float s = 0.f;
  for (int k = threadIdx.x; k < H_SZ; k += 256) s += h[(size_t)b * H_SZ + k] * W_halt[k];
  for (int off = 16; off > 0; off >>= 1) s += __shfl_down(s, off);  // wave32 reduce
  if (lane == 0) red[wave] = s;
  __syncthreads();
  if (threadIdx.x == 0) {
    float tot = 0.f;
    for (int w = 0; w < 8; ++w) tot += red[w];
    float y   = sigm(tot + b_halt[0]);
    float hpo = hp[b];
    float hpn = hpo + y * (1.0f - hpo);
    hp[b]  = hpn;
    rem[b] += (1.0f - hpn);
    nup[b] += 1.0f;
    if (hpn <= THRESH) atomicOr(&flags[1], 1);
  }
}

__global__ void flag_finalize(int* flags) {
  if (flags[0] && !flags[1]) flags[0] = 0;   // active &= (min(hp) <= THRESH)
  flags[1] = 0;
}

// ---------------------------------------------------------------------------
// outputs: avg = hsum * rem / nup  [B,H]; then hp [B]; then nup [B]
// ---------------------------------------------------------------------------
__global__ void write_out(const float* hsum, const float* hp, const float* rem,
                          const float* nup, float* out) {
  int i = blockIdx.x * blockDim.x + threadIdx.x;
  if (i < B_SZ * H_SZ) {
    int b = i / H_SZ;
    out[i] = hsum[i] * rem[b] / nup[b];
  }
  if (i < B_SZ) {
    out[B_SZ * H_SZ + i]        = hp[i];
    out[B_SZ * H_SZ + B_SZ + i] = nup[i];
  }
}

// ---------------------------------------------------------------------------
extern "C" void kernel_launch(void* const* d_in, const int* in_sizes, int n_in,
                              void* d_out, int out_size, void* d_ws, size_t ws_size,
                              hipStream_t stream) {
  const float* x      = (const float*)d_in[0];
  const float* W_ih   = (const float*)d_in[1];
  const float* W_hh   = (const float*)d_in[2];
  const float* b_ih   = (const float*)d_in[3];
  const float* b_hh   = (const float*)d_in[4];
  const float* W_halt = (const float*)d_in[5];
  const float* b_halt = (const float*)d_in[6];
  float* out = (float*)d_out;

  char* ws = (char*)d_ws;
  size_t off = 0;
  auto alloc = [&](size_t bytes) { char* p = ws + off; off += (bytes + 255) & ~(size_t)255; return p; };
  unsigned short* Wih_bf = (unsigned short*)alloc((size_t)G4 * I_SZ * 2);   // 32 MB
  unsigned short* Whh_bf = (unsigned short*)alloc((size_t)G4 * H_SZ * 2);   // 32 MB
  unsigned short* x_bf   = (unsigned short*)alloc((size_t)B_SZ * I_SZ * 2); //  4 MB
  float* xg   = (float*)alloc((size_t)B_SZ * G4 * 4);                       // 32 MB
  float* hbuf = (float*)alloc((size_t)B_SZ * H_SZ * 4);                     //  8 MB
  float* cbuf = (float*)alloc((size_t)B_SZ * H_SZ * 4);                     //  8 MB
  unsigned short* hbf0 = (unsigned short*)alloc((size_t)B_SZ * H_SZ * 2);   //  4 MB
  unsigned short* hbf1 = (unsigned short*)alloc((size_t)B_SZ * H_SZ * 2);   //  4 MB
  float* hsum = (float*)alloc((size_t)B_SZ * H_SZ * 4);                     //  8 MB
  float* hp   = (float*)alloc(B_SZ * 4);
  float* rem  = (float*)alloc(B_SZ * 4);
  float* nup  = (float*)alloc(B_SZ * 4);
  int*   flags = (int*)alloc(64);
  (void)ws_size; (void)in_sizes; (void)n_in; (void)out_size;

  const int T = 256;
  // 1) fp32 -> bf16 conversions (4 elems/thread)
  cvt_bf16<<<(G4 * I_SZ / 4 + T - 1) / T, T, 0, stream>>>(W_ih, Wih_bf, G4 * I_SZ / 4);
  cvt_bf16<<<(G4 * H_SZ / 4 + T - 1) / T, T, 0, stream>>>(W_hh, Whh_bf, G4 * H_SZ / 4);
  cvt_bf16<<<(B_SZ * I_SZ / 4 + T - 1) / T, T, 0, stream>>>(x, x_bf, B_SZ * I_SZ / 4);

  // 2) state init
  init_state<<<(B_SZ * H_SZ + T - 1) / T, T, 0, stream>>>(hbuf, cbuf, hbf0, hsum,
                                                          hp, rem, nup, flags);

  // 3) xg = x @ W_ih^T + b_ih + b_hh  (WMMA)
  gemm_xg<<<dim3(G4 / 64, B_SZ / 64), T, 0, stream>>>(x_bf, Wih_bf, b_ih, b_hh, xg);

  // 4) ACT loop: fixed 10 iterations, device-flag gated (matches scan semantics)
  for (int s = 0; s < NSTEPS; ++s) {
    unsigned short* hin  = (s & 1) ? hbf1 : hbf0;
    unsigned short* hout = (s & 1) ? hbf0 : hbf1;
    lstm_step<<<dim3(H_SZ / 32, B_SZ / 64), T, 0, stream>>>(hin, Whh_bf, xg, cbuf, hbuf,
                                                            hout, hsum, flags);
    halt_update<<<B_SZ, T, 0, stream>>>(hbuf, W_halt, b_halt, hp, rem, nup, flags);
    flag_finalize<<<1, 1, 0, stream>>>(flags);
  }

  // 5) outputs
  write_out<<<(B_SZ * H_SZ + T - 1) / T, T, 0, stream>>>(hsum, hp, rem, nup, out);
}
